// ProductionTPA_73667279061334
// MI455X (gfx1250) — matrix operand order repaired and checked by reading, not compiled, not run on hardware
//
#include <hip/hip_runtime.h>
#include <hip/hip_bf16.h>
#include <math.h>

// ---------- CDNA5 WMMA types ----------
typedef __attribute__((ext_vector_type(16))) __bf16 v16bf;
typedef __attribute__((ext_vector_type(8)))  __bf16 v8bf;
typedef __attribute__((ext_vector_type(4)))  __bf16 v4bf;
typedef __attribute__((ext_vector_type(8)))  float  v8f;

#define NB     8        // batch
#define NT     4096     // tokens
#define ND     1024     // model dim
#define NP     16       // prototypes
#define NH     8        // heads
#define HD     128      // head dim
#define NTOK   (NB*NT)  // 32768
#define NBH    (NB*NH)  // 64
#define TEMP_INV (1.0f/0.07f)

static __device__ __forceinline__ v8f vzero8() {
  v8f z;
#pragma unroll
  for (int i = 0; i < 8; ++i) z[i] = 0.0f;
  return z;
}

static __device__ __forceinline__ v8f wmma_bf16(v16bf a, v16bf b, v8f c) {
  return __builtin_amdgcn_wmma_f32_16x16x32_bf16(false, a, false, b, (short)0, c, false, false);
}

// A-fragment, 16x32 bf16: lanes 0-15 hold K {0..15},
// lanes 16-31 hold K {16..31} of row (lane&15). `rowk` points at
// row start + kBase; `half` = lane>>4.
static __device__ __forceinline__ v16bf load_afrag(const __bf16* __restrict__ rowk, int half) {
  v8bf lo = *(const v8bf*)(rowk + half * 16);
  v8bf hi = *(const v8bf*)(rowk + half * 16 + 8);
  return __builtin_shufflevector(lo, hi, 0,1,2,3,4,5,6,7,8,9,10,11,12,13,14,15);
}

// B-fragment, 32x16 bf16: lane holds column (lane&15), 16 consecutive K
// starting at kBase + (lane>>4)*16. `p` points at those 16 contiguous bf16.
static __device__ __forceinline__ v16bf load_bfrag(const __bf16* __restrict__ p) {
  v8bf lo = *(const v8bf*)(p);
  v8bf hi = *(const v8bf*)(p + 8);
  return __builtin_shufflevector(lo, hi, 0,1,2,3,4,5,6,7,8,9,10,11,12,13,14,15);
}

// ---------- 1) f32 -> bf16 conversion (weights) ----------
__global__ void cvt_bf16_kernel(const float* __restrict__ src, __bf16* __restrict__ dst, int n) {
  int i = blockIdx.x * 256 + threadIdx.x;
  if (i < n) dst[i] = (__bf16)src[i];
}

// ---------- 2) LayerNorm -> bf16 ----------
__global__ __launch_bounds__(256)
void ln_kernel(const float* __restrict__ x, const float* __restrict__ g,
               const float* __restrict__ bb, __bf16* __restrict__ xn) {
  __shared__ float red[256];
  const int row = blockIdx.x;
  const int tid = threadIdx.x;
  const float* xr = x + (size_t)row * ND;
  float v[4];
  float s = 0.0f;
#pragma unroll
  for (int k = 0; k < 4; ++k) { v[k] = xr[tid + 256 * k]; s += v[k]; }
  red[tid] = s; __syncthreads();
  for (int off = 128; off > 0; off >>= 1) { if (tid < off) red[tid] += red[tid + off]; __syncthreads(); }
  const float mu = red[0] * (1.0f / ND);
  __syncthreads();
  float sq = 0.0f;
#pragma unroll
  for (int k = 0; k < 4; ++k) { float d0 = v[k] - mu; sq += d0 * d0; }
  red[tid] = sq; __syncthreads();
  for (int off = 128; off > 0; off >>= 1) { if (tid < off) red[tid] += red[tid + off]; __syncthreads(); }
  const float rstd = rsqrtf(red[0] * (1.0f / ND) + 1e-5f);
  __bf16* outr = xn + (size_t)row * ND;
#pragma unroll
  for (int k = 0; k < 4; ++k) {
    int i = tid + 256 * k;
    outr[i] = (__bf16)(((v[k] - mu) * rstd) * g[i] + bb[i]);
  }
}

// ---------- 3) fused K/V projection GEMM ----------
// Block = 8 waves, block tile = 128 tokens x 64 features; each wave does a
// 32x32 tile of BOTH K and V (A-frags shared). K -> kh[b,h,t,d] (bf16),
// V -> vt[b,h,d,t] (bf16, transposed store is one v8bf per C-frag).
__global__ __launch_bounds__(256)
void kv_gemm_kernel(const __bf16* __restrict__ xn, const __bf16* __restrict__ wk,
                    const __bf16* __restrict__ wv, __bf16* __restrict__ kh,
                    __bf16* __restrict__ vt) {
  const int lane = threadIdx.x & 31;
  const int wave = threadIdx.x >> 5;
  const int r    = lane & 15;
  const int half = lane >> 4;
  const int tokBase  = blockIdx.x * 128 + (wave & 3) * 32;
  const int featBase = blockIdx.y * 64  + (wave >> 2) * 32;

  v8f ck[2][2], cv[2][2];
#pragma unroll
  for (int i = 0; i < 2; ++i)
#pragma unroll
    for (int j = 0; j < 2; ++j) { ck[i][j] = vzero8(); cv[i][j] = vzero8(); }

  const __bf16* a0base = xn + (size_t)(tokBase + r) * ND;
  const __bf16* a1base = xn + (size_t)(tokBase + 16 + r) * ND;
  const __bf16* bk0base = wk + (size_t)(featBase + r) * ND;
  const __bf16* bk1base = wk + (size_t)(featBase + 16 + r) * ND;
  const __bf16* bv0base = wv + (size_t)(featBase + r) * ND;
  const __bf16* bv1base = wv + (size_t)(featBase + 16 + r) * ND;

  for (int kb = 0; kb < ND; kb += 32) {
    v16bf a0 = load_afrag(a0base + kb, half);
    v16bf a1 = load_afrag(a1base + kb, half);
    const int kcol = kb + half * 16;
    v16bf bk0 = load_bfrag(bk0base + kcol);
    v16bf bk1 = load_bfrag(bk1base + kcol);
    v16bf bv0 = load_bfrag(bv0base + kcol);
    v16bf bv1 = load_bfrag(bv1base + kcol);
    ck[0][0] = wmma_bf16(a0, bk0, ck[0][0]);
    ck[0][1] = wmma_bf16(a0, bk1, ck[0][1]);
    ck[1][0] = wmma_bf16(a1, bk0, ck[1][0]);
    ck[1][1] = wmma_bf16(a1, bk1, ck[1][1]);
    cv[0][0] = wmma_bf16(a0, bv0, cv[0][0]);
    cv[0][1] = wmma_bf16(a0, bv1, cv[0][1]);
    cv[1][0] = wmma_bf16(a1, bv0, cv[1][0]);
    cv[1][1] = wmma_bf16(a1, bv1, cv[1][1]);
    if (kb + 32 < ND) {                       // global_prefetch_b8 of next A chunk
      __builtin_prefetch(a0base + kb + 32, 0, 0);
      __builtin_prefetch(a1base + kb + 32, 0, 0);
    }
  }

  // C/D frag: column N = lane&15, rows M = half*8 + i (ISA 7.12.2)
  const int b   = tokBase >> 12;            // token/4096
  const int tt0 = (tokBase & 4095);
#pragma unroll
  for (int am = 0; am < 2; ++am) {
#pragma unroll
    for (int bn = 0; bn < 2; ++bn) {
      const int feat = featBase + bn * 16 + r;
      const int h  = feat >> 7;
      const int dd = feat & 127;
      const int bh = b * NH + h;
      const int t0 = tt0 + am * 16 + half * 8;
      // K: [bh][t][128]
      __bf16* kr = kh + (size_t)bh * NT * HD;
#pragma unroll
      for (int i = 0; i < 8; ++i)
        kr[(size_t)(t0 + i) * HD + dd] = (__bf16)ck[am][bn][i];
      // V transposed: [bh][d][4096] — 8 consecutive t per lane = one b128 store
      v8bf vvec;
#pragma unroll
      for (int i = 0; i < 8; ++i) vvec[i] = (__bf16)cv[am][bn][i];
      *(v8bf*)(vt + ((size_t)bh * HD + dd) * NT + t0) = vvec;
    }
  }
}

// ---------- 4) Q projection (tiny, f32 scalar) ----------
__global__ void qproj_kernel(const float* __restrict__ proto, const float* __restrict__ wq,
                             float* __restrict__ qp) {
  int idx = blockIdx.x * 256 + threadIdx.x;          // 16*1024
  if (idx >= NP * ND) return;
  int p = idx >> 10, e = idx & 1023;
  const float* pr = proto + (size_t)p * ND;
  const float* wr = wq + (size_t)e * ND;
  float s = 0.0f;
  for (int d = 0; d < ND; ++d) s += pr[d] * wr[d];
  qp[idx] = s;
}

// ---------- 5) Q l2-norm per (p,h), store bf16 qh[h][p][128] ----------
__global__ __launch_bounds__(32)
void qnorm_kernel(const float* __restrict__ qp, __bf16* __restrict__ qh) {
  const int row = blockIdx.x;                        // 128 = p*8+h
  const int lane = threadIdx.x;
  const int p = row >> 3, h = row & 7;
  const float* src = qp + (size_t)p * ND + h * HD + lane * 4;
  float v[4]; float sq = 0.0f;
#pragma unroll
  for (int j = 0; j < 4; ++j) { v[j] = src[j]; sq += v[j] * v[j]; }
#pragma unroll
  for (int m = 16; m >= 1; m >>= 1) sq += __shfl_xor(sq, m, 32);
  const float sc = 1.0f / fmaxf(sqrtf(sq), 1e-12f);
  __bf16* dst = qh + ((size_t)h * NP + p) * HD + lane * 4;
#pragma unroll
  for (int j = 0; j < 4; ++j) dst[j] = (__bf16)(v[j] * sc);
}

// ---------- 6) K l2-norm in place, one wave per (bh,t) row of 128 ----------
__global__ __launch_bounds__(32)
void knorm_kernel(__bf16* __restrict__ kh) {
  const size_t row = blockIdx.x;                     // 64*4096 rows
  const int lane = threadIdx.x;
  __bf16* base = kh + row * HD + lane * 4;
  v4bf v = *(const v4bf*)base;
  float f[4]; float sq = 0.0f;
#pragma unroll
  for (int j = 0; j < 4; ++j) { f[j] = (float)v[j]; sq += f[j] * f[j]; }
#pragma unroll
  for (int m = 16; m >= 1; m >>= 1) sq += __shfl_xor(sq, m, 32);
  const float sc = 1.0f / fmaxf(sqrtf(sq), 1e-12f);
  v4bf o;
#pragma unroll
  for (int j = 0; j < 4; ++j) o[j] = (__bf16)(f[j] * sc);
  *(v4bf*)base = o;
}

// ---------- 7) scores = Qh . Kh^T, WMMA, 16x64 per wave ----------
__global__ __launch_bounds__(32)
void scores_kernel(const __bf16* __restrict__ qh, const __bf16* __restrict__ kh,
                   float* __restrict__ scores) {
  const int bh = blockIdx.x;
  const int h  = bh & 7;
  const int tBase = blockIdx.y * 64;
  const int lane = threadIdx.x;
  const int r = lane & 15, half = lane >> 4;

  v16bf a[4];
  const __bf16* qrow = qh + ((size_t)h * NP + r) * HD;
#pragma unroll
  for (int c = 0; c < 4; ++c) a[c] = load_afrag(qrow + c * 32, half);

  float* orow = scores + (size_t)bh * NP * NT;
#pragma unroll
  for (int nt = 0; nt < 4; ++nt) {
    const int tcol = tBase + nt * 16 + r;
    const __bf16* krow = kh + ((size_t)bh * NT + tcol) * HD + half * 16;
    v8f acc = vzero8();
#pragma unroll
    for (int c = 0; c < 4; ++c)
      acc = wmma_bf16(a[c], load_bfrag(krow + c * 32), acc);
#pragma unroll
    for (int i = 0; i < 8; ++i)
      orow[(size_t)(half * 8 + i) * NT + tcol] = acc[i];
  }
}

// ---------- 8) softmax over T (with 1/TEMP), -> bf16 attn ----------
__global__ __launch_bounds__(256)
void softmax_kernel(const float* __restrict__ scores, __bf16* __restrict__ attn) {
  __shared__ float red[256];
  const int row = blockIdx.x;                        // 64*16 rows
  const int tid = threadIdx.x;
  const float* src = scores + (size_t)row * NT;
  float v[16]; float m = -3.0e38f;
#pragma unroll
  for (int k = 0; k < 16; ++k) { v[k] = src[tid + 256 * k]; m = fmaxf(m, v[k]); }
  red[tid] = m; __syncthreads();
  for (int off = 128; off > 0; off >>= 1) { if (tid < off) red[tid] = fmaxf(red[tid], red[tid + off]); __syncthreads(); }
  m = red[0]; __syncthreads();
  float e[16]; float s = 0.0f;
#pragma unroll
  for (int k = 0; k < 16; ++k) { e[k] = __expf((v[k] - m) * TEMP_INV); s += e[k]; }
  red[tid] = s; __syncthreads();
  for (int off = 128; off > 0; off >>= 1) { if (tid < off) red[tid] += red[tid + off]; __syncthreads(); }
  const float inv = 1.0f / red[0];
  __bf16* dst = attn + (size_t)row * NT;
#pragma unroll
  for (int k = 0; k < 16; ++k) dst[tid + 256 * k] = (__bf16)(e[k] * inv);
}

// ---------- 9) pt = attn . V, WMMA over K=4096 ----------
__global__ __launch_bounds__(32)
void pt_kernel(const __bf16* __restrict__ attn, const __bf16* __restrict__ vt,
               float* __restrict__ pt) {
  const int bh = blockIdx.x;
  const int ntile = blockIdx.y;                      // 4 tiles of 32 features
  const int lane = threadIdx.x;
  const int r = lane & 15, half = lane >> 4;

  v8f c0 = vzero8(), c1 = vzero8();
  const __bf16* arow  = attn + ((size_t)bh * NP + r) * NT;
  const __bf16* v0row = vt + ((size_t)bh * HD + ntile * 32 + r) * NT + half * 16;
  const __bf16* v1row = vt + ((size_t)bh * HD + ntile * 32 + 16 + r) * NT + half * 16;
  for (int kb = 0; kb < NT; kb += 32) {
    v16bf a  = load_afrag(arow + kb, half);
    c0 = wmma_bf16(a, load_bfrag(v0row + kb), c0);
    c1 = wmma_bf16(a, load_bfrag(v1row + kb), c1);
  }
  float* out = pt + (size_t)bh * NP * HD;
#pragma unroll
  for (int i = 0; i < 8; ++i) {
    const int p = half * 8 + i;
    out[(size_t)p * HD + ntile * 32 + r]      = c0[i];
    out[(size_t)p * HD + ntile * 32 + 16 + r] = c1[i];
  }
}

// ---------- 10) mean over P -> z_tpa[b][1024] ----------
__global__ void zmean_kernel(const float* __restrict__ pt, float* __restrict__ z) {
  int idx = blockIdx.x * 256 + threadIdx.x;          // 8*1024
  if (idx >= NB * ND) return;
  int b = idx >> 10, e = idx & 1023;
  int h = e >> 7, dd = e & 127;
  const float* base = pt + (size_t)(b * NH + h) * NP * HD + dd;
  float s = 0.0f;
#pragma unroll
  for (int p = 0; p < NP; ++p) s += base[p * HD];
  z[idx] = s * (1.0f / NP);
}

// ---------- 11) MLP (tiny, f32 scalar) ----------
__global__ void mlp1_kernel(const float* __restrict__ z, const float* __restrict__ w1,
                            const float* __restrict__ b1, float* __restrict__ hm) {
  int idx = blockIdx.x * 256 + threadIdx.x;
  if (idx >= NB * ND) return;
  int b = idx >> 10, e = idx & 1023;
  const float* zr = z + (size_t)b * ND;
  const float* wr = w1 + (size_t)e * ND;
  float s = b1[e];
  for (int d = 0; d < ND; ++d) s += zr[d] * wr[d];
  hm[idx] = s / (1.0f + __expf(-s));                 // SiLU
}

__global__ void mlp2_kernel(const float* __restrict__ hm, const float* __restrict__ w2,
                            const float* __restrict__ b2, float* __restrict__ out) {
  int idx = blockIdx.x * 256 + threadIdx.x;
  if (idx >= NB * ND) return;
  int b = idx >> 10, e = idx & 1023;
  const float* hr = hm + (size_t)b * ND;
  const float* wr = w2 + (size_t)e * ND;
  float s = b2[e];
  for (int d = 0; d < ND; ++d) s += hr[d] * wr[d];
  out[idx] = s;
}

// ---------- launch ----------
extern "C" void kernel_launch(void* const* d_in, const int* in_sizes, int n_in,
                              void* d_out, int out_size, void* d_ws, size_t ws_size,
                              hipStream_t stream) {
  const float* x     = (const float*)d_in[0];
  const float* proto = (const float*)d_in[1];
  const float* ln_g  = (const float*)d_in[2];
  const float* ln_b  = (const float*)d_in[3];
  const float* Wq    = (const float*)d_in[4];
  const float* Wk    = (const float*)d_in[5];
  const float* Wv    = (const float*)d_in[6];
  const float* W1    = (const float*)d_in[7];
  const float* b1    = (const float*)d_in[8];
  const float* W2    = (const float*)d_in[9];
  const float* b2    = (const float*)d_in[10];
  float* out = (float*)d_out;

  char* ws = (char*)d_ws;
  size_t off = 0;
  auto alloc = [&](size_t bytes) -> void* {
    void* p = ws + off;
    off = (off + bytes + 255) & ~((size_t)255);
    return p;
  };
  __bf16* xn     = (__bf16*)alloc((size_t)NTOK * ND * 2);       // 67 MB
  __bf16* wkb    = (__bf16*)alloc((size_t)ND * ND * 2);         // 2 MB
  __bf16* wvb    = (__bf16*)alloc((size_t)ND * ND * 2);         // 2 MB
  __bf16* kh     = (__bf16*)alloc((size_t)NBH * NT * HD * 2);   // 67 MB
  __bf16* vt     = (__bf16*)alloc((size_t)NBH * HD * NT * 2);   // 67 MB
  float*  qp     = (float*) alloc((size_t)NP * ND * 4);
  __bf16* qh     = (__bf16*)alloc((size_t)NH * NP * HD * 2);
  float*  scores = (float*) alloc((size_t)NBH * NP * NT * 4);   // 16.8 MB
  __bf16* attn   = (__bf16*)alloc((size_t)NBH * NP * NT * 2);   // 8.4 MB
  float*  pt     = (float*) alloc((size_t)NBH * NP * HD * 4);
  float*  z      = (float*) alloc((size_t)NB * ND * 4);
  float*  hm     = (float*) alloc((size_t)NB * ND * 4);

  cvt_bf16_kernel<<<(ND * ND + 255) / 256, 256, 0, stream>>>(Wk, wkb, ND * ND);
  cvt_bf16_kernel<<<(ND * ND + 255) / 256, 256, 0, stream>>>(Wv, wvb, ND * ND);
  ln_kernel<<<NTOK, 256, 0, stream>>>(x, ln_g, ln_b, xn);
  kv_gemm_kernel<<<dim3(NTOK / 128, ND / 64), 256, 0, stream>>>(xn, wkb, wvb, kh, vt);
  qproj_kernel<<<(NP * ND + 255) / 256, 256, 0, stream>>>(proto, Wq, qp);
  qnorm_kernel<<<NP * NH, 32, 0, stream>>>(qp, qh);
  knorm_kernel<<<NBH * NT, 32, 0, stream>>>(kh);
  scores_kernel<<<dim3(NBH, NT / 64), 32, 0, stream>>>(qh, kh, scores);
  softmax_kernel<<<NBH * NP, 256, 0, stream>>>(scores, attn);
  pt_kernel<<<dim3(NBH, HD / 32), 32, 0, stream>>>(attn, vt, pt);
  zmean_kernel<<<(NB * ND + 255) / 256, 256, 0, stream>>>(pt, z);
  mlp1_kernel<<<(NB * ND + 255) / 256, 256, 0, stream>>>(z, W1, b1, hm);
  mlp2_kernel<<<(NB * ND + 255) / 256, 256, 0, stream>>>(hm, W2, b2, out);
}
